// MultiHeadAttention_70557722739418
// MI455X (gfx1250) — matrix operand order repaired and checked by reading, not compiled, not run on hardware
//
#include <hip/hip_runtime.h>
#include <hip/hip_bf16.h>
#include <stdint.h>

#define B_  4
#define S_  2048
#define D_  512
#define H_  8
#define HD_ 64
#define M_  (B_ * S_)   /* 8192 tokens */
#define N3_ (3 * D_)    /* 1536 */

typedef __bf16 bf16_t;
typedef __attribute__((ext_vector_type(16))) __bf16 v16bf;
typedef __attribute__((ext_vector_type(8)))  float  v8f;

union ABFrag { v16bf v; uint32_t u[8]; };

__device__ __forceinline__ v8f vzero8() {
  v8f z;
#pragma unroll
  for (int i = 0; i < 8; ++i) z[i] = 0.0f;
  return z;
}

// A-matrix fragment (16 rows x 32 K). src points at (row0,k0) of a row-major
// [row][K] bf16 array with leading dim ld (elements, even). ISA layout:
// lane r=lane&15 holds row r; VGPR g holds K = 16*(g>>2) + 8*(lane>>4) + 2*(g&3) + {0,1}.
__device__ __forceinline__ v16bf load_fragA(const bf16_t* src, int ld) {
  const int lane = threadIdx.x & 31;
  const int r = lane & 15;
  const int h = lane >> 4;
  const uint32_t* p = (const uint32_t*)src;
  ABFrag f;
#pragma unroll
  for (int g = 0; g < 8; ++g) {
    int k = ((g >> 2) << 4) + (h << 3) + ((g & 3) << 1);
    f.u[g] = p[(r * ld + k) >> 1];
  }
  return f.v;
}

// B-matrix fragment (32 K x 16 cols). src points at (col0,k0) of a row-major
// [col][K] (i.e. N-major / transposed) bf16 array, leading dim ld. ISA layout:
// lane n=lane&15 holds col n; VGPR v holds K = 16*(v>>2) + 8*(lane>>4) + 2*(v&3) + {0,1}.
__device__ __forceinline__ v16bf load_fragB(const bf16_t* src, int ld) {
  const int lane = threadIdx.x & 31;
  const int n = lane & 15;
  const int h = lane >> 4;
  const uint32_t* p = (const uint32_t*)src;
  ABFrag f;
#pragma unroll
  for (int g = 0; g < 8; ++g) {
    int k = ((g >> 2) << 4) + (h << 3) + ((g & 3) << 1);
    f.u[g] = p[(n * ld + k) >> 1];
  }
  return f.v;
}

__device__ __forceinline__ v8f wmma_bf16(v16bf a, v16bf b, v8f c) {
  return __builtin_amdgcn_wmma_f32_16x16x32_bf16(
      /*neg_a=*/false, a, /*neg_b=*/false, b,
      /*c_mod=*/(short)0, c, /*reuse_a=*/false, /*reuse_b=*/false);
}

// ---------------------------------------------------------------------------
// fp32 -> bf16 conversion, optionally transposing (dst becomes [cols][rows]).
__global__ void convert_bf16_kernel(const float* __restrict__ src,
                                    bf16_t* __restrict__ dst,
                                    int rows, int cols, int transpose) {
  int idx = blockIdx.x * blockDim.x + threadIdx.x;
  if (idx >= rows * cols) return;
  int r = idx / cols, c = idx % cols;
  float v = src[idx];
  dst[transpose ? (c * rows + r) : idx] = (bf16_t)v;
}

// ---------------------------------------------------------------------------
// QKV projection: [8192,512] x [512,1536] + bias, scattered into
// Q [b,h,s,64], K [b,h,s,64], Vt [b,h,64,s] (V transposed), all bf16.
// One wave computes a 32(M) x 64(N) tile.
__global__ void qkv_gemm_kernel(const bf16_t* __restrict__ xb,
                                const bf16_t* __restrict__ wT,   // [1536][512]
                                const float* __restrict__ bias,  // [1536]
                                bf16_t* __restrict__ Qb,
                                bf16_t* __restrict__ Kb,
                                bf16_t* __restrict__ Vt) {
  const int wave = blockIdx.x * (blockDim.x >> 5) + (threadIdx.x >> 5);
  const int m0 = (wave & 255) * 32;   // 8192/32 = 256 M tiles
  const int n0 = (wave >> 8) * 64;    // 1536/64 = 24  N tiles

  v8f acc[2][4];
#pragma unroll
  for (int mt = 0; mt < 2; ++mt)
#pragma unroll
    for (int nt = 0; nt < 4; ++nt) acc[mt][nt] = vzero8();

  for (int k0 = 0; k0 < D_; k0 += 32) {
    v16bf a0 = load_fragA(xb + (m0 + 0)  * D_ + k0, D_);
    v16bf a1 = load_fragA(xb + (m0 + 16) * D_ + k0, D_);
#pragma unroll
    for (int nt = 0; nt < 4; ++nt) {
      v16bf b = load_fragB(wT + (n0 + nt * 16) * D_ + k0, D_);
      acc[0][nt] = wmma_bf16(a0, b, acc[0][nt]);
      acc[1][nt] = wmma_bf16(a1, b, acc[1][nt]);
    }
  }

  const int lane = threadIdx.x & 31;
  const int cl = lane & 15, h = lane >> 4;
#pragma unroll
  for (int mt = 0; mt < 2; ++mt) {
#pragma unroll
    for (int nt = 0; nt < 4; ++nt) {
      int n = n0 + nt * 16 + cl;
      float bv = bias[n];
      int head = n / 192;
      int wn = n % 192;
      int chunk = wn / 64;
      int col = wn & 63;
#pragma unroll
      for (int i = 0; i < 8; ++i) {
        int m = m0 + mt * 16 + i + 8 * h;
        float c = acc[mt][nt][i] + bv;
        int b = m >> 11, s = m & (S_ - 1);
        int bh = b * H_ + head;
        if (chunk == 0)      Qb[(bh * S_ + s) * HD_ + col] = (bf16_t)c;
        else if (chunk == 1) Kb[(bh * S_ + s) * HD_ + col] = (bf16_t)c;
        else                 Vt[(bh * HD_ + col) * S_ + s] = (bf16_t)c;
      }
    }
  }
}

// ---------------------------------------------------------------------------
// Flash attention: one wave per (b,h, 16-query block). Online softmax over
// 32-key blocks; scores never touch global memory.
__global__ void attention_kernel(const bf16_t* __restrict__ Qb,
                                 const bf16_t* __restrict__ Kb,
                                 const bf16_t* __restrict__ Vt,
                                 bf16_t* __restrict__ aob) {
  __shared__ bf16_t psc[4][16 * 32];   // per-wave P re-layout scratch (1 KB each)
  const int wslot = threadIdx.x >> 5;
  const int wave = blockIdx.x * 4 + wslot;
  const int qb = wave & 127;           // 2048/16 = 128 query blocks
  const int bh = wave >> 7;            // 0..31
  const int q0 = qb * 16;
  const int lane = threadIdx.x & 31;
  const int cl = lane & 15, h = lane >> 4;

  const bf16_t* Qp = Qb + (size_t)bh * S_ * HD_;
  const bf16_t* Kp = Kb + (size_t)bh * S_ * HD_;
  const bf16_t* Vp = Vt + (size_t)bh * HD_ * S_;

  v16bf qa0 = load_fragA(Qp + q0 * HD_ + 0, HD_);
  v16bf qa1 = load_fragA(Qp + q0 * HD_ + 32, HD_);

  v8f o[4];
#pragma unroll
  for (int t = 0; t < 4; ++t) o[t] = vzero8();
  float mrow[8], lrow[8];
#pragma unroll
  for (int i = 0; i < 8; ++i) { mrow[i] = -1e30f; lrow[i] = 0.0f; }

  const int qlast = q0 + 15;
  for (int key0 = 0; key0 <= qlast; key0 += 32) {
    // ---- S = Q K^T (16 x 32 tile, fp32) ----
    v8f s0 = vzero8(), s1 = vzero8();
    {
      v16bf kb0 = load_fragB(Kp + key0 * HD_ + 0, HD_);
      v16bf kb1 = load_fragB(Kp + (key0 + 16) * HD_ + 0, HD_);
      s0 = wmma_bf16(qa0, kb0, s0);
      s1 = wmma_bf16(qa0, kb1, s1);
      kb0 = load_fragB(Kp + key0 * HD_ + 32, HD_);
      kb1 = load_fragB(Kp + (key0 + 16) * HD_ + 32, HD_);
      s0 = wmma_bf16(qa1, kb0, s0);
      s1 = wmma_bf16(qa1, kb1, s1);
    }

    // ---- scale + causal mask + online softmax ----
#pragma unroll
    for (int i = 0; i < 8; ++i) {
      int row = q0 + i + 8 * h;
      float a = s0[i] * 0.125f;
      float b = s1[i] * 0.125f;
      if (key0 + cl > row)      a = -1e30f;
      if (key0 + 16 + cl > row) b = -1e30f;
      float mx = fmaxf(a, b);
#pragma unroll
      for (int msk = 1; msk <= 8; msk <<= 1)
        mx = fmaxf(mx, __shfl_xor(mx, msk, 32));   // stays within 16-lane half
      float mnew = fmaxf(mrow[i], mx);
      float alpha = __expf(mrow[i] - mnew);
      mrow[i] = mnew;
      float e0 = __expf(a - mnew);
      float e1 = __expf(b - mnew);
      float rs = e0 + e1;
#pragma unroll
      for (int msk = 1; msk <= 8; msk <<= 1)
        rs += __shfl_xor(rs, msk, 32);
      lrow[i] = lrow[i] * alpha + rs;
#pragma unroll
      for (int t = 0; t < 4; ++t) o[t][i] *= alpha;
      // P (bf16) into wave-private LDS, [row][key] row-major, ld = 32
      psc[wslot][(i + 8 * h) * 32 + cl]      = (bf16_t)e0;
      psc[wslot][(i + 8 * h) * 32 + 16 + cl] = (bf16_t)e1;
    }

    // ---- O += P V (LDS ops from same wave are in-order) ----
    v16bf pa = load_fragA(&psc[wslot][0], 32);
#pragma unroll
    for (int t = 0; t < 4; ++t) {
      v16bf vb = load_fragB(Vp + (t * 16) * S_ + key0, S_);
      o[t] = wmma_bf16(pa, vb, o[t]);
    }
  }

  // ---- normalize + store attention output as bf16 tokens x D ----
  const int b = bh >> 3, head = bh & 7;
#pragma unroll
  for (int t = 0; t < 4; ++t) {
#pragma unroll
    for (int i = 0; i < 8; ++i) {
      int s = q0 + i + 8 * h;
      float val = o[t][i] / lrow[i];
      aob[((b * S_ + s) * D_) + head * HD_ + t * 16 + cl] = (bf16_t)val;
    }
  }
}

// ---------------------------------------------------------------------------
// Output projection: [8192,512] x [512,512] + bias -> fp32 out.
__global__ void out_proj_kernel(const bf16_t* __restrict__ aob,
                                const bf16_t* __restrict__ woT,  // [512][512]
                                const float* __restrict__ bo,
                                float* __restrict__ out) {
  const int wave = blockIdx.x * (blockDim.x >> 5) + (threadIdx.x >> 5);
  const int m0 = (wave & 255) * 32;   // 256 M tiles
  const int n0 = (wave >> 8) * 64;    // 8   N tiles

  v8f acc[2][4];
#pragma unroll
  for (int mt = 0; mt < 2; ++mt)
#pragma unroll
    for (int nt = 0; nt < 4; ++nt) acc[mt][nt] = vzero8();

  for (int k0 = 0; k0 < D_; k0 += 32) {
    v16bf a0 = load_fragA(aob + (m0 + 0)  * D_ + k0, D_);
    v16bf a1 = load_fragA(aob + (m0 + 16) * D_ + k0, D_);
#pragma unroll
    for (int nt = 0; nt < 4; ++nt) {
      v16bf b = load_fragB(woT + (n0 + nt * 16) * D_ + k0, D_);
      acc[0][nt] = wmma_bf16(a0, b, acc[0][nt]);
      acc[1][nt] = wmma_bf16(a1, b, acc[1][nt]);
    }
  }

  const int lane = threadIdx.x & 31;
  const int cl = lane & 15, h = lane >> 4;
#pragma unroll
  for (int mt = 0; mt < 2; ++mt) {
#pragma unroll
    for (int nt = 0; nt < 4; ++nt) {
      int n = n0 + nt * 16 + cl;
      float bv = bo[n];
#pragma unroll
      for (int i = 0; i < 8; ++i) {
        int m = m0 + mt * 16 + i + 8 * h;
        out[m * D_ + n] = acc[mt][nt][i] + bv;
      }
    }
  }
}

// ---------------------------------------------------------------------------
extern "C" void kernel_launch(void* const* d_in, const int* in_sizes, int n_in,
                              void* d_out, int out_size, void* d_ws, size_t ws_size,
                              hipStream_t stream) {
  const float* x    = (const float*)d_in[0];
  // d_in[1] = additive mask: unused, causal mask computed analytically
  const float* Wqkv = (const float*)d_in[2];
  const float* bqkv = (const float*)d_in[3];
  const float* Wo   = (const float*)d_in[4];
  const float* bo   = (const float*)d_in[5];
  float* out = (float*)d_out;

  char* ws = (char*)d_ws;
  bf16_t* xb    = (bf16_t*)(ws + 0);           // 8192*512  bf16 =  8 MiB
  bf16_t* wqkvT = (bf16_t*)(ws + 8388608);     // 1536*512  bf16 = 1.5 MiB
  bf16_t* woT   = (bf16_t*)(ws + 9961472);     // 512*512   bf16 = 0.5 MiB
  bf16_t* Qb    = (bf16_t*)(ws + 10485760);    // [b,h,s,64]      =  8 MiB
  bf16_t* Kb    = (bf16_t*)(ws + 18874368);    // [b,h,s,64]      =  8 MiB
  bf16_t* Vt    = (bf16_t*)(ws + 27262976);    // [b,h,64,s]      =  8 MiB
  bf16_t* aob   = (bf16_t*)(ws + 35651584);    // [tokens,512]    =  8 MiB

  convert_bf16_kernel<<<(M_ * D_ + 255) / 256, 256, 0, stream>>>(x, xb, M_, D_, 0);
  convert_bf16_kernel<<<(D_ * N3_ + 255) / 256, 256, 0, stream>>>(Wqkv, wqkvT, D_, N3_, 1);
  convert_bf16_kernel<<<(D_ * D_ + 255) / 256, 256, 0, stream>>>(Wo, woT, D_, D_, 1);

  // 256 M-tiles * 24 N-tiles = 6144 waves / 4 waves-per-block
  qkv_gemm_kernel<<<1536, 128, 0, stream>>>(xb, wqkvT, bqkv, Qb, Kb, Vt);
  // 32 (b,h) * 128 q-blocks = 4096 waves / 4
  attention_kernel<<<1024, 128, 0, stream>>>(Qb, Kb, Vt, aob);
  // 256 M-tiles * 8 N-tiles = 2048 waves / 4
  out_proj_kernel<<<512, 128, 0, stream>>>(aob, woT, bo, out);
}
